// Head_5480378270207
// MI455X (gfx1250) — compile-verified, matrix operand order checked by reading, not run on hardware
//
#include <hip/hip_runtime.h>

#define Ed 384
#define Td 256
#define Hd 64
#define KSTR 72    // halves; 144B row stride for lK/lQ (multiple of 16B, conflict-breaking)
#define VSTR 264   // halves; 528B row stride for lVt
#define EPAD 392   // halves; 784B row stride for staged W^T

typedef __attribute__((ext_vector_type(16))) _Float16 v16h;
typedef __attribute__((ext_vector_type(8)))  _Float16 v8h;
typedef __attribute__((ext_vector_type(8)))  float    v8f;

__device__ __forceinline__ v8f wmma_f16(v16h a, v16h b, v8f c) {
    // D = A(16x32 f16) * B(32x16 f16) + C(16x16 f32)
    return __builtin_amdgcn_wmma_f32_16x16x32_f16(
        /*neg_a=*/false, a, /*neg_b=*/false, b,
        /*c_mod=*/(short)0, c, /*reuse_a=*/false, /*reuse_b=*/false);
}

__device__ __forceinline__ v8f v8f_zero() {
    v8f z = {0.f, 0.f, 0.f, 0.f, 0.f, 0.f, 0.f, 0.f};
    return z;
}

// v16h A/B fragment from LDS: elements 0..7 at p, 8..15 at p+16 halves.
// p must already include the +fhi*8 half offset; 16B alignment guaranteed by layout.
__device__ __forceinline__ v16h frag_lds(const _Float16* p) {
    v16h f;
    *(v8h*)&f       = *(const v8h*)p;
    *((v8h*)&f + 1) = *(const v8h*)(p + 16);
    return f;
}

// One projection: dst = x @ W, W staged in LDS as W^T f16, result to LDS.
// MAT is compile-time so destination addressing is branch-free straight-line code.
template <int MAT>
__device__ __forceinline__ void project(const float* __restrict__ W,
                                        const float* __restrict__ xb,
                                        _Float16* lW, _Float16* lK, _Float16* lQ,
                                        _Float16* lVt,
                                        int wave, int l15, int fhi) {
    // stage W^T into LDS as f16 (coalesced b128 global reads)
    for (int idx = threadIdx.x * 4; idx < Ed * Hd; idx += 256 * 4) {
        const float4 w4 = *(const float4*)(W + idx);
        const int e = idx >> 6;        // E index
        const int h = idx & 63;        // H index (multiple of 4)
        lW[(h + 0) * EPAD + e] = (_Float16)w4.x;
        lW[(h + 1) * EPAD + e] = (_Float16)w4.y;
        lW[(h + 2) * EPAD + e] = (_Float16)w4.z;
        lW[(h + 3) * EPAD + e] = (_Float16)w4.w;
    }
    __syncthreads();

    #pragma unroll
    for (int rt = 0; rt < 2; ++rt) {
        const int m0 = wave * 32 + rt * 16;   // token row-tile base
        v8f acc[4];
        #pragma unroll
        for (int ct = 0; ct < 4; ++ct) acc[ct] = v8f_zero();

        for (int kc = 0; kc < 12; ++kc) {
            const int e0 = kc * 32;
            // A fragment: lane(l15,fhi) -> x row m0+l15, k = {0..7,16..23}+8*fhi
            const float4* xr4 = (const float4*)(xb + (size_t)(m0 + l15) * Ed + e0 + fhi * 8);
            const float4 a0 = xr4[0], a1 = xr4[1], a2 = xr4[4], a3 = xr4[5];
            v16h af;
            af[0]  = (_Float16)a0.x; af[1]  = (_Float16)a0.y; af[2]  = (_Float16)a0.z; af[3]  = (_Float16)a0.w;
            af[4]  = (_Float16)a1.x; af[5]  = (_Float16)a1.y; af[6]  = (_Float16)a1.z; af[7]  = (_Float16)a1.w;
            af[8]  = (_Float16)a2.x; af[9]  = (_Float16)a2.y; af[10] = (_Float16)a2.z; af[11] = (_Float16)a2.w;
            af[12] = (_Float16)a3.x; af[13] = (_Float16)a3.y; af[14] = (_Float16)a3.z; af[15] = (_Float16)a3.w;

            #pragma unroll
            for (int ct = 0; ct < 4; ++ct) {
                // B fragment from W^T: lane carries column h = ct*16+l15
                const v16h bf = frag_lds(lW + (ct * 16 + l15) * EPAD + e0 + fhi * 8);
                acc[ct] = wmma_f16(af, bf, acc[ct]);
            }
        }
        // D layout: lane(l15,fhi) elem r -> row m0 + r + 8*fhi, col ct*16 + l15
        #pragma unroll
        for (int ct = 0; ct < 4; ++ct) {
            const int h0 = ct * 16 + l15;
            if (MAT == 2) {
                // V^T: consecutive r -> consecutive t along lVt row: one b128 store
                v8h pk;
                #pragma unroll
                for (int r = 0; r < 8; ++r) pk[r] = (_Float16)acc[ct][r];
                *(v8h*)(lVt + h0 * VSTR + m0 + fhi * 8) = pk;
            } else {
                _Float16* dst = (MAT == 0) ? lK : lQ;
                #pragma unroll
                for (int r = 0; r < 8; ++r)
                    dst[(m0 + fhi * 8 + r) * KSTR + h0] = (_Float16)acc[ct][r];
            }
        }
    }
    __syncthreads();   // compute done before next W staging / phase 2
}

__global__ __launch_bounds__(256, 1)
void head_attn_kernel(const float* __restrict__ x,
                      const float* __restrict__ Wk,
                      const float* __restrict__ Wq,
                      const float* __restrict__ Wv,
                      float* __restrict__ out) {
    __shared__ _Float16 lK[Td * KSTR];   // K row-major [t][h]
    __shared__ _Float16 lQ[Td * KSTR];   // Q row-major [t][h]
    __shared__ _Float16 lVt[Hd * VSTR];  // V transposed [h][t]
    __shared__ _Float16 lW[Hd * EPAD];   // current W transposed [h][e] (f16)

    const int b    = blockIdx.x;
    const int lane = threadIdx.x & 31;
    const int wave = threadIdx.x >> 5;
    const int fhi  = lane >> 4;    // half-wave select (k-offset +8 in A/B fragments)
    const int l15  = lane & 15;

    const float* xb = x + (size_t)b * Td * Ed;

    // ========== Phase 1: K,Q,V = x @ W, one matrix at a time (W staged in LDS) ==========
    project<0>(Wk, xb, lW, lK, lQ, lVt, wave, l15, fhi);
    project<1>(Wq, xb, lW, lK, lQ, lVt, wave, l15, fhi);
    project<2>(Wv, xb, lW, lK, lQ, lVt, wave, l15, fhi);

    // ============ Phase 2: causal flash attention via S^T = K @ Q^T ============
    // Lane(l15,*) owns query t = t0 + l15; its 8 elems are s-rows r + 8*fhi.
    // Causal load balance: wave w takes query tiles {w, 15-w} -> every wave does
    // exactly 9 key-pair iterations (vs 2..16 with the naive 2w/2w+1 split).
    float* ob = out + (size_t)b * Td * Hd;

    for (int qt = 0; qt < 2; ++qt) {
        const int tile = (qt == 0) ? wave : (15 - wave);
        const int t0 = tile * 16;
        const int tq = t0 + l15;

        // Q^T B-fragments (reused across all key tiles): lane carries column t.
        const v16h qf0 = frag_lds(lQ + tq * KSTR +  0 + fhi * 8);
        const v16h qf1 = frag_lds(lQ + tq * KSTR + 32 + fhi * 8);

        v8f o[4];
        #pragma unroll
        for (int ht = 0; ht < 4; ++ht) o[ht] = v8f_zero();
        float mrow = 0.f, lrow = 0.f;

        const int npairs = (t0 + 47) >> 5;   // key s-pairs (32 keys each), causal bound
        for (int p = 0; p < npairs; ++p) {
            const int s0 = p * 32;
            v8f s[2] = { v8f_zero(), v8f_zero() };
            #pragma unroll
            for (int st = 0; st < 2; ++st) {
                const _Float16* kr = lK + (s0 + st * 16 + l15) * KSTR + fhi * 8;
                s[st] = wmma_f16(frag_lds(kr),      qf0, s[st]);
                s[st] = wmma_f16(frag_lds(kr + 32), qf1, s[st]);
            }
            // scale 1/sqrt(64), causal mask, per-lane tile max
            float tmax = -1e30f;
            #pragma unroll
            for (int st = 0; st < 2; ++st)
                #pragma unroll
                for (int r = 0; r < 8; ++r) {
                    const int sa = s0 + st * 16 + fhi * 8 + r;
                    float v = s[st][r] * 0.125f;
                    v = (sa <= tq) ? v : -1e30f;
                    s[st][r] = v;
                    tmax = fmaxf(tmax, v);
                }
            // merge max across the lane pair sharing this query (xor 16)
            tmax = fmaxf(tmax, __shfl_xor(tmax, 16, 32));
            const float mnew = fmaxf(mrow, tmax);
            const float corr = __expf(mrow - mnew);
            mrow = mnew;

            // exp; D-layout of s[] is exactly the P^T B-fragment layout:
            // element st*8+r <-> k = st*16 + r + 8*fhi
            v16h pf;
            float psum = 0.f;
            #pragma unroll
            for (int st = 0; st < 2; ++st)
                #pragma unroll
                for (int r = 0; r < 8; ++r) {
                    const float pv = __expf(s[st][r] - mnew);
                    psum += pv;
                    pf[st * 8 + r] = (_Float16)pv;
                }
            lrow = lrow * corr + psum;

            // O^T[h][t] += V^T @ P^T  (A = V^T rows h, k = s)
            #pragma unroll
            for (int ht = 0; ht < 4; ++ht) {
                #pragma unroll
                for (int r = 0; r < 8; ++r) o[ht][r] *= corr;
                const _Float16* vr = lVt + (ht * 16 + l15) * VSTR + s0 + fhi * 8;
                o[ht] = wmma_f16(frag_lds(vr), pf, o[ht]);
            }
        }
        const float ltot = lrow + __shfl_xor(lrow, 16, 32);
        const float inv  = 1.f / ltot;
        // O^T D layout: lane(l15,fhi) elem r -> h = ht*16 + r + 8*fhi, t = t0 + l15
        float4* o4 = (float4*)(ob + (size_t)tq * Hd);
        #pragma unroll
        for (int ht = 0; ht < 4; ++ht) {
            float4 lo, hi4;
            lo.x  = o[ht][0] * inv; lo.y  = o[ht][1] * inv;
            lo.z  = o[ht][2] * inv; lo.w  = o[ht][3] * inv;
            hi4.x = o[ht][4] * inv; hi4.y = o[ht][5] * inv;
            hi4.z = o[ht][6] * inv; hi4.w = o[ht][7] * inv;
            o4[ht * 4 + fhi * 2]     = lo;
            o4[ht * 4 + fhi * 2 + 1] = hi4;
        }
    }
}

extern "C" void kernel_launch(void* const* d_in, const int* in_sizes, int n_in,
                              void* d_out, int out_size, void* d_ws, size_t ws_size,
                              hipStream_t stream) {
    (void)in_sizes; (void)n_in; (void)out_size; (void)d_ws; (void)ws_size;
    const float* x  = (const float*)d_in[0];
    const float* Wk = (const float*)d_in[1];
    const float* Wq = (const float*)d_in[2];
    const float* Wv = (const float*)d_in[3];
    float* out = (float*)d_out;
    head_attn_kernel<<<dim3(512), dim3(256), 0, stream>>>(x, Wk, Wq, Wv, out);
}
